// TemporalAttention_78958678769949
// MI455X (gfx1250) — matrix operand order repaired and checked by reading, not compiled, not run on hardware
//
#include <hip/hip_runtime.h>
#include <hip/hip_bf16.h>

typedef __bf16 bf16_t;
typedef __attribute__((ext_vector_type(16))) __bf16 v16bf;
typedef __attribute__((ext_vector_type(8)))  float  v8f;
typedef unsigned int v4u __attribute__((ext_vector_type(4)));
typedef int          v8i __attribute__((ext_vector_type(8)));
typedef int          v4i __attribute__((ext_vector_type(4)));

#define EMBED    512
#define INTER    512
#define HEADS    8
#define HEAD_DIM 64
#define TSEQ     512
#define PP       16
#define BB       2
#define NROWS    (BB * TSEQ * PP)   // 16384

#define V8F_ZERO {0.f,0.f,0.f,0.f,0.f,0.f,0.f,0.f}

__device__ __forceinline__ v16bf load_frag16(const bf16_t* p0, const bf16_t* p1) {
  v16bf r;
  ((uint4*)&r)[0] = *(const uint4*)p0;   // 16B load
  ((uint4*)&r)[1] = *(const uint4*)p1;   // 16B load
  return r;
}

__device__ __forceinline__ v8f wmma_bf16(v16bf a, v16bf b, v8f c) {
  // D = A(16x32 bf16) x B(32x16 bf16) + C(16x16 f32)
  return __builtin_amdgcn_wmma_f32_16x16x32_bf16(false, a, false, b, (short)0, c, false, false);
}

// TDM: stage a contiguous 16-row x 512-col bf16 strip (16 KB) into LDS.
// D# group0: count=1, lds_addr, 57-bit global_addr, type=2 ("image").
// D# group1: data_size=1 (2B), tensor 512x16, tile 512x16, dim0 stride 512.
// Remaining groups zero (tile_dim2/3 == 0 -> unused).
// This toolchain exposes the 6-arg builtin: (v4u, v8i, v4i, v4i, v8i, cpol).
__device__ __forceinline__ void tdm_load_strip(bf16_t* lds_dst, const bf16_t* gsrc) {
  const unsigned long long ga = (unsigned long long)(uintptr_t)gsrc;
  const unsigned int ldsa = (unsigned int)(uintptr_t)lds_dst;   // low 32 bits = LDS offset
  v4u g0 = { 1u,                                  // count=1, user descriptor
             ldsa,                                // lds_addr
             (unsigned int)ga,                    // global_addr[31:0]
             ((unsigned int)(ga >> 32) & 0x01FFFFFFu) | 0x80000000u }; // addr[56:32] | type=2
  v8i g1 = { (int)0x00010000u,                    // data_size=1 (2 bytes)
             (int)(512u << 16),                   // tensor_dim0[15:0]=512
             (int)(16u  << 16),                   // tensor_dim1[15:0]=16
             (int)(512u << 16),                   // tile_dim0=512
             16,                                  // tile_dim1=16
             512,                                 // tensor_dim0_stride[31:0]=512
             0, 0 };
  v4i gz4 = { 0, 0, 0, 0 };
  v8i gz8 = { 0, 0, 0, 0, 0, 0, 0, 0 };
  __builtin_amdgcn_tensor_load_to_lds(g0, g1, gz4, gz4, gz8, 0);
}

// ---------------------------------------------------------------------------
// Kernel 0: fp32 -> bf16 conversion of activations and weights
// ---------------------------------------------------------------------------
__global__ __launch_bounds__(256) void convert_kernel(
    const float* __restrict__ x,  const float* __restrict__ qw,
    const float* __restrict__ kw, const float* __restrict__ vw,
    const float* __restrict__ pw,
    bf16_t* __restrict__ xbf, bf16_t* __restrict__ qwb, bf16_t* __restrict__ kwb,
    bf16_t* __restrict__ vwb, bf16_t* __restrict__ pwb) {
  const int i = blockIdx.x * 256 + threadIdx.x;       // grid == NROWS*EMBED/256
  xbf[i] = (bf16_t)x[i];
  if (i < INTER * EMBED) {
    qwb[i] = (bf16_t)qw[i];
    kwb[i] = (bf16_t)kw[i];
    vwb[i] = (bf16_t)vw[i];
    pwb[i] = (bf16_t)pw[i];
  }
}

// ---------------------------------------------------------------------------
// Kernel 1: fused QKV GEMM + bias + RoPE, scatter to head-major layouts.
// Block = one 16-row A-strip (TDM-staged in LDS) x 8 N-tiles (one per wave).
// 1024 mtiles x 4 blocks-per-mtile = 4096 blocks.
// ---------------------------------------------------------------------------
__global__ __launch_bounds__(256) void qkv_rope_kernel(
    const bf16_t* __restrict__ xbf,
    const bf16_t* __restrict__ qw, const bf16_t* __restrict__ kw,
    const bf16_t* __restrict__ vw,
    const float* __restrict__ qb, const float* __restrict__ kbias,
    const float* __restrict__ vb,
    bf16_t* __restrict__ Qo, bf16_t* __restrict__ Ko, bf16_t* __restrict__ Vt) {
  __shared__ __align__(16) bf16_t Asmem[16 * EMBED];      // 16 KB A strip
  const int lane     = threadIdx.x & 31;
  const int wave     = threadIdx.x >> 5;
  const int laneLo   = lane & 15;
  const int laneHalf = lane >> 4;
  const int mtile    = blockIdx.x >> 2;                   // 0..1023
  const int ntile    = ((blockIdx.x & 3) << 3) | wave;    // 0..31
  const int colbase  = ntile << 4;

  if (wave == 0) {                                        // one TDM op per block
    tdm_load_strip(Asmem, xbf + (size_t)(mtile << 4) * EMBED);
    __builtin_amdgcn_s_wait_tensorcnt(0);
  }
  __syncthreads();

  const bf16_t* arow = Asmem + (size_t)laneLo * EMBED;    // LDS A-fragments
  const bf16_t* qwp  = qw + (size_t)(colbase + laneLo) * EMBED + laneHalf * 16;
  const bf16_t* kwp  = kw + (size_t)(colbase + laneLo) * EMBED + laneHalf * 16;
  const bf16_t* vwp  = vw + (size_t)(colbase + laneLo) * EMBED + laneHalf * 16;

  v8f cq = V8F_ZERO, ck = V8F_ZERO, cv = V8F_ZERO;
#pragma unroll 4
  for (int kc = 0; kc < 16; ++kc) {       // K = 512 in chunks of 32
    const bf16_t* ap = arow + kc * 32 + laneHalf * 16;
    v16bf a = load_frag16(ap, ap + 8);   // ds_load_b128 x2
    cq = wmma_bf16(a, load_frag16(qwp + kc * 32, qwp + kc * 32 + 8), cq);
    ck = wmma_bf16(a, load_frag16(kwp + kc * 32, kwp + kc * 32 + 8), ck);
    cv = wmma_bf16(a, load_frag16(vwp + kc * 32, vwp + kc * 32 + 8), cv);
  }

  // Epilogue: bias + RoPE + scatter.  Tile rows == p = 0..15 for fixed (b,t).
  const int t    = mtile & (TSEQ - 1);
  const int bidx = mtile >> 9;
  const int col  = colbase + laneLo;      // 0..511 within INTER
  const int h    = col >> 6;
  const int d    = col & 63;
  const float fr = (float)t * __powf(10000.0f, -(float)(d & ~1) * (1.0f / 64.0f));
  const float cc = __cosf(fr), ss = __sinf(fr);
  const float bqv = qb[col], bkv = kbias[col], bvv = vb[col];
#pragma unroll
  for (int v = 0; v < 8; ++v) {
    const int m = v + (laneHalf << 3);    // tile row == p index
    float xq = cq[v] + bqv;
    float xk = ck[v] + bkv;
    float xv = cv[v] + bvv;
    float xq_p = __shfl_xor(xq, 1, 32);   // RoPE pair partner (d ^ 1)
    float xk_p = __shfl_xor(xk, 1, 32);
    float rq = (d & 1) ? (xq * cc + xq_p * ss) : (xq * cc - xq_p * ss);
    float rk = (d & 1) ? (xk * cc + xk_p * ss) : (xk * cc - xk_p * ss);
    const size_t qidx = (((size_t)(bidx * PP + m) * HEADS + h) * TSEQ + t) * HEAD_DIM + d;
    Qo[qidx] = (bf16_t)rq;
    Ko[qidx] = (bf16_t)rk;
    const size_t vidx = (((size_t)(bidx * PP + m) * HEADS + h) * HEAD_DIM + d) * TSEQ + t;
    Vt[vidx] = (bf16_t)xv;               // V stored transposed [head][d][t]
  }
}

// ---------------------------------------------------------------------------
// Kernel 2: causal flash attention.  One wave = 16 query rows of one head.
// 8 waves/block -> 128 query rows; 4 blocks per head; 256 heads -> 1024 blocks.
// ---------------------------------------------------------------------------
__global__ __launch_bounds__(256) void flash_attn_kernel(
    const bf16_t* __restrict__ Qm, const bf16_t* __restrict__ Km,
    const bf16_t* __restrict__ Vtm, bf16_t* __restrict__ Y) {
  __shared__ __align__(16) bf16_t lds[8 * 16 * 32];   // 8 KB, wave-private strips
  const int lane     = threadIdx.x & 31;
  const int wave     = threadIdx.x >> 5;
  const int laneLo   = lane & 15;
  const int laneHalf = lane >> 4;
  const int head     = blockIdx.x >> 2;               // 0..255
  const int qblock   = ((((blockIdx.x & 3) << 3) + wave) << 4);
  const int bidx     = head >> 7;
  const int p        = (head >> 3) & 15;
  const int h        = head & 7;

  const bf16_t* Qh = Qm  + (size_t)head * TSEQ * HEAD_DIM;
  const bf16_t* Kh = Km  + (size_t)head * TSEQ * HEAD_DIM;
  const bf16_t* Vh = Vtm + (size_t)head * HEAD_DIM * TSEQ;

  // Q A-fragments for the two d-chunks (d=0..31 and d=32..63), kept resident.
  const bf16_t* qp = Qh + (size_t)(qblock + laneLo) * HEAD_DIM + laneHalf * 16;
  const v16bf aq0 = load_frag16(qp,      qp + 8);
  const v16bf aq1 = load_frag16(qp + 32, qp + 40);

  v8f o0 = V8F_ZERO, o1 = V8F_ZERO, o2 = V8F_ZERO, o3 = V8F_ZERO;
  float mrun[8], lrun[8];
#pragma unroll
  for (int v = 0; v < 8; ++v) { mrun[v] = -__builtin_inff(); lrun[v] = 0.f; }

  bf16_t* Pl = lds + wave * (16 * 32);
  const float scale = 0.125f;                          // 1/sqrt(64)
  const int nkb = (qblock + 47) >> 5;                  // causal-trimmed key blocks

  for (int kb = 0; kb < nkb; ++kb) {
    const int k0 = kb << 5;
    // ---- S = Q K^T, two 16x16 N-tiles, contraction D=64 as 2x K=32 ----
    const bf16_t* kp0 = Kh + (size_t)(k0 + laneLo) * HEAD_DIM + laneHalf * 16;
    v8f s0 = V8F_ZERO, s1 = V8F_ZERO;
    s0 = wmma_bf16(aq0, load_frag16(kp0,      kp0 + 8),  s0);
    s0 = wmma_bf16(aq1, load_frag16(kp0 + 32, kp0 + 40), s0);
    const bf16_t* kp1 = kp0 + 16 * HEAD_DIM;
    s1 = wmma_bf16(aq0, load_frag16(kp1,      kp1 + 8),  s1);
    s1 = wmma_bf16(aq1, load_frag16(kp1 + 32, kp1 + 40), s1);

    // ---- online softmax (rows live across the 16-lane halves) ----
    const int kg0 = k0 + laneLo;
    const int kg1 = kg0 + 16;
#pragma unroll
    for (int v = 0; v < 8; ++v) {
      const int qg = qblock + v + (laneHalf << 3);
      float e0 = (kg0 <= qg) ? s0[v] * scale : -__builtin_inff();
      float e1 = (kg1 <= qg) ? s1[v] * scale : -__builtin_inff();
      float mx = fmaxf(e0, e1);
      mx = fmaxf(mx, __shfl_xor(mx, 1, 16));
      mx = fmaxf(mx, __shfl_xor(mx, 2, 16));
      mx = fmaxf(mx, __shfl_xor(mx, 4, 16));
      mx = fmaxf(mx, __shfl_xor(mx, 8, 16));
      const float mnew = fmaxf(mrun[v], mx);
      const float p0 = __expf(e0 - mnew);
      const float p1 = __expf(e1 - mnew);
      float rs = p0 + p1;
      rs += __shfl_xor(rs, 1, 16);
      rs += __shfl_xor(rs, 2, 16);
      rs += __shfl_xor(rs, 4, 16);
      rs += __shfl_xor(rs, 8, 16);
      const float corr = __expf(mrun[v] - mnew);
      lrun[v] = lrun[v] * corr + rs;
      mrun[v] = mnew;
      o0[v] *= corr; o1[v] *= corr; o2[v] *= corr; o3[v] *= corr;
      const int row = v + (laneHalf << 3);
      Pl[row * 32 + laneLo]      = (bf16_t)p0;   // C-layout -> row-major LDS
      Pl[row * 32 + 16 + laneLo] = (bf16_t)p1;
    }
    asm volatile("s_wait_dscnt 0" ::: "memory");   // wave-private LDS RAW fence

    // reload P as 16x32 A-fragment (ds_load_b128 x2)
    const bf16_t* pp = Pl + laneLo * 32 + laneHalf * 16;
    v16bf ap = load_frag16(pp, pp + 8);

    // ---- O += P V : V^T layout makes B-fragments contiguous ----
    const bf16_t* vp0 = Vh + (size_t)laneLo * TSEQ + k0 + laneHalf * 16;
    o0 = wmma_bf16(ap, load_frag16(vp0, vp0 + 8), o0);
    const bf16_t* vp1 = vp0 + 16 * TSEQ;
    o1 = wmma_bf16(ap, load_frag16(vp1, vp1 + 8), o1);
    const bf16_t* vp2 = vp0 + 32 * TSEQ;
    o2 = wmma_bf16(ap, load_frag16(vp2, vp2 + 8), o2);
    const bf16_t* vp3 = vp0 + 48 * TSEQ;
    o3 = wmma_bf16(ap, load_frag16(vp3, vp3 + 8), o3);
  }

  // ---- normalize and scatter to y[b,t,p, h*64+d] (bf16) ----
#pragma unroll
  for (int v = 0; v < 8; ++v) {
    const float invl = 1.0f / lrun[v];
    const int qg = qblock + v + (laneHalf << 3);
    const size_t yrow = ((size_t)(bidx * TSEQ + qg) * PP + p) * INTER;
    const int cb = h * HEAD_DIM + laneLo;
    Y[yrow + cb]      = (bf16_t)(o0[v] * invl);
    Y[yrow + cb + 16] = (bf16_t)(o1[v] * invl);
    Y[yrow + cb + 32] = (bf16_t)(o2[v] * invl);
    Y[yrow + cb + 48] = (bf16_t)(o3[v] * invl);
  }
}

// ---------------------------------------------------------------------------
// Kernel 3: output projection  out = y @ proj_w^T   (fp32 result)
// Same TDM-staged A-strip structure as kernel 1.
// ---------------------------------------------------------------------------
__global__ __launch_bounds__(256) void proj_kernel(
    const bf16_t* __restrict__ ybf, const bf16_t* __restrict__ pw,
    float* __restrict__ out) {
  __shared__ __align__(16) bf16_t Asmem[16 * INTER];      // 16 KB A strip
  const int lane     = threadIdx.x & 31;
  const int wave     = threadIdx.x >> 5;
  const int laneLo   = lane & 15;
  const int laneHalf = lane >> 4;
  const int mtile    = blockIdx.x >> 2;
  const int ntile    = ((blockIdx.x & 3) << 3) | wave;
  const int colbase  = ntile << 4;

  if (wave == 0) {
    tdm_load_strip(Asmem, ybf + (size_t)(mtile << 4) * INTER);
    __builtin_amdgcn_s_wait_tensorcnt(0);
  }
  __syncthreads();

  const bf16_t* arow = Asmem + (size_t)laneLo * INTER;
  const bf16_t* bwp  = pw + (size_t)(colbase + laneLo) * INTER + laneHalf * 16;

  v8f acc = V8F_ZERO;
#pragma unroll 4
  for (int kc = 0; kc < 16; ++kc) {
    const bf16_t* ap = arow + kc * 32 + laneHalf * 16;
    v16bf a = load_frag16(ap, ap + 8);
    acc = wmma_bf16(a, load_frag16(bwp + kc * 32, bwp + kc * 32 + 8), acc);
  }
#pragma unroll
  for (int v = 0; v < 8; ++v) {
    const size_t r = (size_t)(mtile << 4) + v + (laneHalf << 3);
    out[r * EMBED + colbase + laneLo] = acc[v];
  }
}

// ---------------------------------------------------------------------------
extern "C" void kernel_launch(void* const* d_in, const int* in_sizes, int n_in,
                              void* d_out, int out_size, void* d_ws, size_t ws_size,
                              hipStream_t stream) {
  (void)in_sizes; (void)n_in; (void)out_size; (void)ws_size;
  const float* x  = (const float*)d_in[0];
  const float* qw = (const float*)d_in[1];
  const float* qb = (const float*)d_in[2];
  const float* kw = (const float*)d_in[3];
  const float* kb = (const float*)d_in[4];
  const float* vw = (const float*)d_in[5];
  const float* vb = (const float*)d_in[6];
  const float* pw = (const float*)d_in[7];
  float* out = (float*)d_out;

  const size_t NX = (size_t)NROWS * EMBED;   // 8,388,608
  const size_t NW = (size_t)INTER * EMBED;   //   262,144
  bf16_t* xbf = (bf16_t*)d_ws;
  bf16_t* qwb = xbf + NX;
  bf16_t* kwb = qwb + NW;
  bf16_t* vwb = kwb + NW;
  bf16_t* pwb = vwb + NW;
  bf16_t* Q   = pwb + NW;
  bf16_t* K   = Q + NX;
  bf16_t* Vt  = K + NX;
  bf16_t* Y   = Vt + NX;                     // total ~86 MB of workspace

  convert_kernel<<<(int)(NX / 256), 256, 0, stream>>>(x, qw, kw, vw, pw,
                                                      xbf, qwb, kwb, vwb, pwb);
  qkv_rope_kernel<<<(1024 * 32) / 8, 256, 0, stream>>>(xbf, qwb, kwb, vwb,
                                                       qb, kb, vb, Q, K, Vt);
  flash_attn_kernel<<<256 * 4, 256, 0, stream>>>(Q, K, Vt, Y);
  proj_kernel<<<(1024 * 32) / 8, 256, 0, stream>>>(Y, pwb, out);
}
